// DaLUKE_18691697672597
// MI455X (gfx1250) — matrix operand order repaired and checked by reading, not compiled, not run on hardware
//
#include <hip/hip_runtime.h>
#include <math.h>

typedef _Float16 half_t;
typedef _Float16 v16h_t __attribute__((ext_vector_type(16)));
typedef _Float16 v8h_t  __attribute__((ext_vector_type(8)));
typedef float    v8f_t  __attribute__((ext_vector_type(8)));
typedef int      v4i_t  __attribute__((ext_vector_type(4)));

static constexpr int H   = 768;
static constexpr int NH  = 12;
static constexpr int DH  = 64;
static constexpr int NL  = 4;
static constexpr int FF  = 3072;
static constexpr int B   = 8;
static constexpr int LW  = 512;
static constexpr int LE  = 128;
static constexpr int T   = LW + LE;   // 640
static constexpr int MPOS = 30;
static constexpr int EE  = 256;

static constexpr int BM = 128;
static constexpr int BN = 64;
static constexpr int BK = 32;
static constexpr int LDK = BK + 8;    // 40 halves/row -> 80B stride, keeps 16B chunks aligned

#ifndef __has_builtin
#define __has_builtin(x) 0
#endif
#if __has_builtin(__builtin_amdgcn_global_load_async_to_lds_b128) && \
    __has_builtin(__builtin_amdgcn_s_wait_asynccnt)
#define USE_ASYNC_LDS 1
#else
#define USE_ASYNC_LDS 0
#endif

__device__ __forceinline__ float gelu_exact(float x) {
    return 0.5f * x * (1.0f + erff(x * 0.70710678118654752f));
}

// copy 16 bytes global(f16) -> LDS, async when available
__device__ __forceinline__ void cp16(const half_t* g, half_t* l) {
#if USE_ASYNC_LDS
    __builtin_amdgcn_global_load_async_to_lds_b128((v4i_t*)g, (v4i_t*)l, 0, 0);
#else
    *(v8h_t*)l = *(const v8h_t*)g;
#endif
}

template <int N>
__device__ __forceinline__ void wait_stage() {
#if USE_ASYNC_LDS
    __builtin_amdgcn_s_wait_asynccnt(N);
#endif
}

__device__ __forceinline__ v16h_t frag16(const half_t* p0, const half_t* p1) {
    v8h_t a = *(const v8h_t*)p0;
    v8h_t b = *(const v8h_t*)p1;
    return __builtin_shufflevector(a, b, 0,1,2,3,4,5,6,7,8,9,10,11,12,13,14,15);
}

__device__ __forceinline__ v8f_t wmma16(v16h_t a, v16h_t b, v8f_t c) {
    return __builtin_amdgcn_wmma_f32_16x16x32_f16(false, a, false, b, (short)0, c,
                                                  false, false);
}

// ---------------------------------------------------------------------------
// WMMA GEMM: C[z] = act(A[z](MxK f16, lda) * Bt[z]^T + bias)
//   A  : f16 [M][K]  row-major (lda)
//   Bt : f16 [N][K]  row-major (ldb)   (i.e. B pre-transposed; fragment-contiguous)
//   C  : TOUT; CT=true stores C transposed as [N][M] (ldc = M-stride)
// Block tile 128x64x32, 8 waves, 4 WMMAs/wave/k-step, double-buffered async LDS.
// Requires M%128==0, N%64==0, K%32==0.
// ---------------------------------------------------------------------------
template <bool DOGELU, bool CT, typename TOUT>
__global__ __launch_bounds__(256) void gemm_wmma(
    const half_t* __restrict__ A, int lda, long long zsa,
    const half_t* __restrict__ Bt, int ldb, long long zsb,
    TOUT* __restrict__ C, int ldc, long long zsc,
    const float* __restrict__ bias, int M, int N, int K)
{
    __shared__ __attribute__((aligned(16))) half_t As[2][BM][LDK];
    __shared__ __attribute__((aligned(16))) half_t Bs[2][BN][LDK];

    const int tid = threadIdx.x;
    const int z   = blockIdx.z;
    A  += (long long)z * zsa;
    Bt += (long long)z * zsb;
    C  += (long long)z * zsc;

    const int tM = blockIdx.y * BM;
    const int tN = blockIdx.x * BN;

    const int wid  = tid >> 5;
    const int lane = tid & 31;
    const int lm   = lane & 15;
    const int g    = lane >> 4;

    // A tile: 128 rows x 64B -> 512 chunks of 16B, 2 per thread (rows ar, ar+64)
    const int ar = tid >> 2;
    const int ao = (tid & 3) * 8;      // half offset within row
    // B tile: 64 rows x 64B -> 256 chunks, 1 per thread
    const int br = tid >> 2;
    const int bo = (tid & 3) * 8;

    v8f_t acc[4] = {};

    auto stage = [&](int kk2, int ib2) {
        const half_t* ga = A + (long long)(tM + ar) * lda + kk2 + ao;
        cp16(ga, &As[ib2][ar][ao]);
        cp16(ga + 64ll * lda, &As[ib2][ar + 64][ao]);
        const half_t* gb = Bt + (long long)(tN + br) * ldb + kk2 + bo;
        cp16(gb, &Bs[ib2][br][bo]);
    };

    stage(0, 0);
    int ib = 0;
    for (int kk = 0; kk < K; kk += BK, ib ^= 1) {
        const bool more = (kk + BK) < K;
        if (more) {
            stage(kk + BK, ib ^ 1);   // prefetch next tile (overlaps compute)
            wait_stage<3>();          // oldest 3 asyncs (current tile) done
        } else {
            wait_stage<0>();
        }
        __syncthreads();

        // hoist ALL fragment loads ahead of the WMMAs so ds_loads batch up
        const int am = wid * 16 + lm;
        v16h_t af = frag16(&As[ib][am][g * 8], &As[ib][am][16 + g * 8]);
        v16h_t bf[4];
        #pragma unroll
        for (int s = 0; s < 4; ++s) {
            const int bn = s * 16 + lm;
            bf[s] = frag16(&Bs[ib][bn][g * 8], &Bs[ib][bn][16 + g * 8]);
        }
        #pragma unroll
        for (int s = 0; s < 4; ++s)
            acc[s] = wmma16(af, bf[s], acc[s]);
        __syncthreads();
    }

    // C/D layout: VGPR r -> m = r + 8*g ; lane -> n = lane & 15
    const int crow = tM + wid * 16 + 8 * g;
    #pragma unroll
    for (int s = 0; s < 4; ++s) {
        const int col = tN + s * 16 + lm;
        const float bv = bias ? bias[col] : 0.0f;
        #pragma unroll
        for (int r = 0; r < 8; ++r) {
            float v = acc[s][r] + bv;
            if (DOGELU) v = gelu_exact(v);
            if (CT) C[(long long)col * ldc + (crow + r)] = (TOUT)v;
            else    C[(long long)(crow + r) * ldc + col] = (TOUT)v;
        }
    }
}

// ---------------------------------------------------------------------------
// Transpose + fp32->f16 weight conversion: dst[z][n][k] = (f16)src[z][k][n]
// ---------------------------------------------------------------------------
__global__ __launch_bounds__(256) void wtrans_f16(const float* __restrict__ src,
                                                  half_t* __restrict__ dst,
                                                  int K, int N)
{
    __shared__ half_t tile[32][33];
    const long long z = blockIdx.z;
    src += z * (long long)K * N;
    dst += z * (long long)K * N;
    const int n0 = blockIdx.x * 32, k0 = blockIdx.y * 32;
    const int tx = threadIdx.x & 31, ty = threadIdx.x >> 5;
    #pragma unroll
    for (int i = 0; i < 4; ++i) {
        int k = ty + i * 8;
        tile[k][tx] = (half_t)src[(long long)(k0 + k) * N + n0 + tx];
    }
    __syncthreads();
    #pragma unroll
    for (int i = 0; i < 4; ++i) {
        int n = ty + i * 8;
        dst[(long long)(n0 + n) * K + k0 + tx] = tile[tx][n];
    }
}

// ---------------------------------------------------------------------------
// Word embedding + LayerNorm -> hidden (f32) + hidden_h (f16)
// ---------------------------------------------------------------------------
__global__ __launch_bounds__(256) void embed_word_ln(
    const int* __restrict__ ids, const int* __restrict__ segs,
    const float* __restrict__ wemb, const float* __restrict__ wpos,
    const float* __restrict__ wtyp, const float* __restrict__ gam,
    const float* __restrict__ bet, float* __restrict__ hidden,
    half_t* __restrict__ hidden_h)
{
    const int r = blockIdx.x;            // b*LW + t
    const int b = r / LW, t = r % LW;
    const int tid = threadIdx.x;
    const long long id = ids[r];
    const int sg = segs[r];
    __shared__ float s1[256], s2[256];

    float v[3], sum = 0.f, sq = 0.f;
    #pragma unroll
    for (int i = 0; i < 3; ++i) {
        int c = tid + i * 256;
        float x = wemb[id * H + c] + wpos[(long long)t * H + c] + wtyp[(long long)sg * H + c];
        v[i] = x; sum += x; sq += x * x;
    }
    s1[tid] = sum; s2[tid] = sq; __syncthreads();
    for (int o = 128; o > 0; o >>= 1) {
        if (tid < o) { s1[tid] += s1[tid + o]; s2[tid] += s2[tid + o]; }
        __syncthreads();
    }
    float mean = s1[0] * (1.0f / H);
    float var  = s2[0] * (1.0f / H) - mean * mean;
    float rstd = rsqrtf(var + 1e-12f);
    const long long row = (long long)b * T + t;
    #pragma unroll
    for (int i = 0; i < 3; ++i) {
        int c = tid + i * 256;
        float y = (v[i] - mean) * rstd * gam[c] + bet[c];
        hidden[row * H + c]   = y;
        hidden_h[row * H + c] = (half_t)y;
    }
}

// ---------------------------------------------------------------------------
__global__ __launch_bounds__(256) void ent_gather(
    const int* __restrict__ ids, const float* __restrict__ emb,
    half_t* __restrict__ egath)
{
    const int r = blockIdx.x;
    egath[(long long)r * EE + threadIdx.x] =
        (half_t)emb[(long long)ids[r] * EE + threadIdx.x];
}

// ---------------------------------------------------------------------------
__global__ __launch_bounds__(256) void ent_finish_ln(
    const float* __restrict__ dense, const int* __restrict__ epos,
    const int* __restrict__ esegs, const float* __restrict__ postab,
    const float* __restrict__ etyp, const float* __restrict__ gam,
    const float* __restrict__ bet, float* __restrict__ hidden,
    half_t* __restrict__ hidden_h)
{
    const int r = blockIdx.x;            // b*LE + e
    const int b = r / LE, e = r % LE;
    const int tid = threadIdx.x;
    const int sg = esegs[r];
    const int* pp = epos + (long long)r * MPOS;
    __shared__ float s1[256], s2[256];

    float ps[3] = {0.f, 0.f, 0.f};
    float cnt = 0.f;
    for (int m = 0; m < MPOS; ++m) {
        int p = pp[m];
        if (p >= 0) {
            cnt += 1.0f;
            #pragma unroll
            for (int i = 0; i < 3; ++i)
                ps[i] += postab[(long long)p * H + tid + i * 256];
        }
    }
    float inv = 1.0f / fmaxf(cnt, 1e-7f);

    float v[3], sum = 0.f, sq = 0.f;
    #pragma unroll
    for (int i = 0; i < 3; ++i) {
        int c = tid + i * 256;
        float x = dense[(long long)r * H + c] + ps[i] * inv + etyp[(long long)sg * H + c];
        v[i] = x; sum += x; sq += x * x;
    }
    s1[tid] = sum; s2[tid] = sq; __syncthreads();
    for (int o = 128; o > 0; o >>= 1) {
        if (tid < o) { s1[tid] += s1[tid + o]; s2[tid] += s2[tid + o]; }
        __syncthreads();
    }
    float mean = s1[0] * (1.0f / H);
    float var  = s2[0] * (1.0f / H) - mean * mean;
    float rstd = rsqrtf(var + 1e-12f);
    const long long row = (long long)b * T + LW + e;
    #pragma unroll
    for (int i = 0; i < 3; ++i) {
        int c = tid + i * 256;
        float y = (v[i] - mean) * rstd * gam[c] + bet[c];
        hidden[row * H + c]   = y;
        hidden_h[row * H + c] = (half_t)y;
    }
}

// ---------------------------------------------------------------------------
// Masked softmax: reads f32 scores[NH][T][T], writes f16 probs
// ---------------------------------------------------------------------------
__global__ __launch_bounds__(256) void softmax_rows(
    const float* __restrict__ scores, half_t* __restrict__ probs_h,
    const int* __restrict__ wmask, const int* __restrict__ emask, float scale)
{
    const int q = blockIdx.x, h = blockIdx.y, tid = threadIdx.x;
    const long long rowoff = ((long long)h * T + q) * T;
    const float* row = scores + rowoff;
    half_t* out = probs_h + rowoff;
    __shared__ float red[256];

    float v[3];
    float mx = -3.0e38f;
    #pragma unroll
    for (int i = 0; i < 3; ++i) {
        int k = tid + i * 256;
        if (k < T) {
            int mv = (k < LW) ? wmask[k] : emask[k - LW];
            float x = row[k] * scale + 10000.0f * ((float)mv - 1.0f);
            v[i] = x; mx = fmaxf(mx, x);
        } else v[i] = -3.0e38f;
    }
    red[tid] = mx; __syncthreads();
    for (int o = 128; o > 0; o >>= 1) {
        if (tid < o) red[tid] = fmaxf(red[tid], red[tid + o]);
        __syncthreads();
    }
    mx = red[0]; __syncthreads();

    float sum = 0.f;
    #pragma unroll
    for (int i = 0; i < 3; ++i) {
        int k = tid + i * 256;
        if (k < T) { v[i] = expf(v[i] - mx); sum += v[i]; }
    }
    red[tid] = sum; __syncthreads();
    for (int o = 128; o > 0; o >>= 1) {
        if (tid < o) red[tid] += red[tid + o];
        __syncthreads();
    }
    float inv = 1.0f / red[0];
    #pragma unroll
    for (int i = 0; i < 3; ++i) {
        int k = tid + i * 256;
        if (k < T) out[k] = (half_t)(v[i] * inv);
    }
}

// ---------------------------------------------------------------------------
// out = LayerNorm(x + res); writes f32 + f16 copies
// ---------------------------------------------------------------------------
__global__ __launch_bounds__(256) void add_res_ln(
    const float* __restrict__ x, const float* __restrict__ res,
    const float* __restrict__ gam, const float* __restrict__ bet,
    float* __restrict__ outf, half_t* __restrict__ outh)
{
    const long long r = blockIdx.x;
    const int tid = threadIdx.x;
    __shared__ float s1[256], s2[256];
    float v[3], sum = 0.f, sq = 0.f;
    #pragma unroll
    for (int i = 0; i < 3; ++i) {
        int c = tid + i * 256;
        float a = x[r * H + c] + res[r * H + c];
        v[i] = a; sum += a; sq += a * a;
    }
    s1[tid] = sum; s2[tid] = sq; __syncthreads();
    for (int o = 128; o > 0; o >>= 1) {
        if (tid < o) { s1[tid] += s1[tid + o]; s2[tid] += s2[tid + o]; }
        __syncthreads();
    }
    float mean = s1[0] * (1.0f / H);
    float var  = s2[0] * (1.0f / H) - mean * mean;
    float rstd = rsqrtf(var + 1e-12f);
    #pragma unroll
    for (int i = 0; i < 3; ++i) {
        int c = tid + i * 256;
        float y = (v[i] - mean) * rstd * gam[c] + bet[c];
        outf[r * H + c] = y;
        outh[r * H + c] = (half_t)y;
    }
}

// ---------------------------------------------------------------------------
__global__ __launch_bounds__(256) void copy_out_kernel(
    const float* __restrict__ hidden, float* __restrict__ out)
{
    long long idx = (long long)blockIdx.x * 256 + threadIdx.x;
    int c = (int)(idx % H);
    long long th = idx / H;
    int t = (int)(th % T);
    int b = (int)(th / T);
    long long dst = (t < LW)
        ? ((long long)(b * LW + t) * H + c)
        : ((long long)B * LW * H + (long long)(b * LE + (t - LW)) * H + c);
    out[dst] = hidden[idx];
}

// ---------------------------------------------------------------------------
extern "C" void kernel_launch(void* const* d_in, const int* in_sizes, int n_in,
                              void* d_out, int out_size, void* d_ws, size_t ws_size,
                              hipStream_t stream) {
    (void)in_sizes; (void)n_in; (void)out_size; (void)ws_size;

    const int* word_ids      = (const int*)d_in[0];
    const int* word_segments = (const int*)d_in[1];
    const int* word_mask     = (const int*)d_in[2];
    const int* entity_ids    = (const int*)d_in[3];
    const int* entity_pos    = (const int*)d_in[4];
    const int* entity_segs   = (const int*)d_in[5];
    const int* entity_mask   = (const int*)d_in[6];

    const float* word_emb    = (const float*)d_in[7];
    const float* word_pos    = (const float*)d_in[8];
    const float* word_typ    = (const float*)d_in[9];
    const float* we_ln_g     = (const float*)d_in[10];
    const float* we_ln_b     = (const float*)d_in[11];
    const float* ent_emb     = (const float*)d_in[12];
    const float* ent_dense_w = (const float*)d_in[13];
    const float* ent_dense_b = (const float*)d_in[14];
    const float* ent_pos_t   = (const float*)d_in[15];
    const float* ent_typ     = (const float*)d_in[16];
    const float* ee_ln_g     = (const float*)d_in[17];
    const float* ee_ln_b     = (const float*)d_in[18];
    const float* Qw_w   = (const float*)d_in[19];
    const float* Qw_b   = (const float*)d_in[20];
    const float* Qe_w   = (const float*)d_in[21];
    const float* Qe_b   = (const float*)d_in[22];
    const float* Qw2e_w = (const float*)d_in[23];
    const float* Qw2e_b = (const float*)d_in[24];
    const float* Qe2w_w = (const float*)d_in[25];
    const float* Qe2w_b = (const float*)d_in[26];
    const float* K_w    = (const float*)d_in[27];
    const float* K_b    = (const float*)d_in[28];
    const float* V_w    = (const float*)d_in[29];
    const float* V_b    = (const float*)d_in[30];
    const float* so_w   = (const float*)d_in[31];
    const float* so_b   = (const float*)d_in[32];
    const float* so_ln_g = (const float*)d_in[33];
    const float* so_ln_b = (const float*)d_in[34];
    const float* int_w   = (const float*)d_in[35];
    const float* int_b   = (const float*)d_in[36];
    const float* out_w   = (const float*)d_in[37];
    const float* out_b   = (const float*)d_in[38];
    const float* out_ln_g = (const float*)d_in[39];
    const float* out_ln_b = (const float*)d_in[40];

    char* base = (char*)d_ws;
    size_t off = 0;
    auto allocf = [&](size_t n) { float* p = (float*)(base + off);
                                  off += (n * 4 + 15) & ~15ull; return p; };
    auto alloch = [&](size_t n) { half_t* p = (half_t*)(base + off);
                                  off += (n * 2 + 15) & ~15ull; return p; };

    // fp32 buffers
    float* hidden   = allocf((size_t)B * T * H);
    float* scoresf  = allocf((size_t)NH * T * T);
    float* sobuf    = allocf((size_t)B * T * H);
    float* ff2      = allocf((size_t)B * T * H);
    float* entdense = allocf((size_t)B * LE * H);
    // f16 activations
    half_t* hidden_h = alloch((size_t)B * T * H);
    half_t* qw_h     = alloch((size_t)B * LW * H);
    half_t* qw2e_h   = alloch((size_t)B * LW * H);
    half_t* qe_h     = alloch((size_t)B * LE * H);
    half_t* qe2w_h   = alloch((size_t)B * LE * H);
    half_t* k_h      = alloch((size_t)B * T * H);
    half_t* vt_h     = alloch((size_t)B * H * T);     // V transposed per batch: [H][T]
    half_t* ctx_h    = alloch((size_t)B * T * H);
    half_t* so_h     = alloch((size_t)B * T * H);
    half_t* ff_h     = alloch((size_t)B * T * FF);
    half_t* probs_h  = alloch((size_t)NH * T * T);
    half_t* egath_h  = alloch((size_t)B * LE * EE);
    // f16 transposed weights [N][K] (per layer)
    half_t* qwT   = alloch((size_t)NL * H * H);
    half_t* qeT   = alloch((size_t)NL * H * H);
    half_t* qw2eT = alloch((size_t)NL * H * H);
    half_t* qe2wT = alloch((size_t)NL * H * H);
    half_t* kT    = alloch((size_t)NL * H * H);
    half_t* vT    = alloch((size_t)NL * H * H);
    half_t* soT   = alloch((size_t)NL * H * H);
    half_t* intT  = alloch((size_t)NL * H * FF);
    half_t* outT  = alloch((size_t)NL * FF * H);
    half_t* entdT = alloch((size_t)EE * H);

    // ---- one-time weight transpose+f16 conversion ----
    {
        dim3 ghh(H / 32, H / 32, NL);
        wtrans_f16<<<ghh, 256, 0, stream>>>(Qw_w,   qwT,   H, H);
        wtrans_f16<<<ghh, 256, 0, stream>>>(Qe_w,   qeT,   H, H);
        wtrans_f16<<<ghh, 256, 0, stream>>>(Qw2e_w, qw2eT, H, H);
        wtrans_f16<<<ghh, 256, 0, stream>>>(Qe2w_w, qe2wT, H, H);
        wtrans_f16<<<ghh, 256, 0, stream>>>(K_w,    kT,    H, H);
        wtrans_f16<<<ghh, 256, 0, stream>>>(V_w,    vT,    H, H);
        wtrans_f16<<<ghh, 256, 0, stream>>>(so_w,   soT,   H, H);
        wtrans_f16<<<dim3(FF / 32, H / 32, NL), 256, 0, stream>>>(int_w, intT, H, FF);
        wtrans_f16<<<dim3(H / 32, FF / 32, NL), 256, 0, stream>>>(out_w, outT, FF, H);
        wtrans_f16<<<dim3(H / 32, EE / 32, 1), 256, 0, stream>>>(ent_dense_w, entdT, EE, H);
    }

    // ---- Embeddings ----
    embed_word_ln<<<B * LW, 256, 0, stream>>>(word_ids, word_segments, word_emb,
                                              word_pos, word_typ, we_ln_g, we_ln_b,
                                              hidden, hidden_h);
    ent_gather<<<B * LE, 256, 0, stream>>>(entity_ids, ent_emb, egath_h);
    gemm_wmma<false, false, float><<<dim3(H / BN, (B * LE) / BM, 1), 256, 0, stream>>>(
        egath_h, EE, 0, entdT, EE, 0, entdense, H, 0, ent_dense_b, B * LE, H, EE);
    ent_finish_ln<<<B * LE, 256, 0, stream>>>(entdense, entity_pos, entity_segs,
                                              ent_pos_t, ent_typ, ee_ln_g, ee_ln_b,
                                              hidden, hidden_h);

    const float scale = 1.0f / sqrtf((float)NH);

    for (int l = 0; l < NL; ++l) {
        const half_t* qwT_l   = qwT   + (size_t)l * H * H;
        const half_t* qeT_l   = qeT   + (size_t)l * H * H;
        const half_t* qw2eT_l = qw2eT + (size_t)l * H * H;
        const half_t* qe2wT_l = qe2wT + (size_t)l * H * H;
        const half_t* kT_l    = kT    + (size_t)l * H * H;
        const half_t* vT_l    = vT    + (size_t)l * H * H;
        const half_t* soT_l   = soT   + (size_t)l * H * H;
        const half_t* intT_l  = intT  + (size_t)l * H * FF;
        const half_t* outT_l  = outT  + (size_t)l * FF * H;

        // ---- Q projections (Z = batch; word/entity row slices of hidden) ----
        gemm_wmma<false, false, half_t><<<dim3(H / BN, LW / BM, B), 256, 0, stream>>>(
            hidden_h, H, (long long)T * H, qwT_l, H, 0,
            qw_h, H, (long long)LW * H, Qw_b + l * H, LW, H, H);
        gemm_wmma<false, false, half_t><<<dim3(H / BN, LW / BM, B), 256, 0, stream>>>(
            hidden_h, H, (long long)T * H, qw2eT_l, H, 0,
            qw2e_h, H, (long long)LW * H, Qw2e_b + l * H, LW, H, H);
        gemm_wmma<false, false, half_t><<<dim3(H / BN, LE / BM, B), 256, 0, stream>>>(
            hidden_h + (long long)LW * H, H, (long long)T * H, qeT_l, H, 0,
            qe_h, H, (long long)LE * H, Qe_b + l * H, LE, H, H);
        gemm_wmma<false, false, half_t><<<dim3(H / BN, LE / BM, B), 256, 0, stream>>>(
            hidden_h + (long long)LW * H, H, (long long)T * H, qe2wT_l, H, 0,
            qe2w_h, H, (long long)LE * H, Qe2w_b + l * H, LE, H, H);
        // K: flat rows [B*T][H]
        gemm_wmma<false, false, half_t><<<dim3(H / BN, (B * T) / BM, 1), 256, 0, stream>>>(
            hidden_h, H, 0, kT_l, H, 0, k_h, H, 0, K_b + l * H, B * T, H, H);
        // V: per batch, stored transposed [H][T] for fragment-contiguous PV B-operand
        gemm_wmma<false, true, half_t><<<dim3(H / BN, T / BM, B), 256, 0, stream>>>(
            hidden_h, H, (long long)T * H, vT_l, H, 0,
            vt_h, T, (long long)H * T, V_b + l * H, T, H, H);

        // ---- Attention per batch (Z = heads) ----
        for (int b = 0; b < B; ++b) {
            const half_t* kb = k_h + (long long)b * T * H;
            // word -> word
            gemm_wmma<false, false, float><<<dim3(LW / BN, LW / BM, NH), 256, 0, stream>>>(
                qw_h + (long long)b * LW * H, H, DH, kb, H, DH,
                scoresf, T, (long long)T * T, nullptr, LW, LW, DH);
            // word -> entity
            gemm_wmma<false, false, float><<<dim3(LE / BN, LW / BM, NH), 256, 0, stream>>>(
                qw2e_h + (long long)b * LW * H, H, DH, kb + (long long)LW * H, H, DH,
                scoresf + LW, T, (long long)T * T, nullptr, LW, LE, DH);
            // entity -> word
            gemm_wmma<false, false, float><<<dim3(LW / BN, LE / BM, NH), 256, 0, stream>>>(
                qe2w_h + (long long)b * LE * H, H, DH, kb, H, DH,
                scoresf + (long long)LW * T, T, (long long)T * T, nullptr, LE, LW, DH);
            // entity -> entity
            gemm_wmma<false, false, float><<<dim3(LE / BN, LE / BM, NH), 256, 0, stream>>>(
                qe_h + (long long)b * LE * H, H, DH, kb + (long long)LW * H, H, DH,
                scoresf + (long long)LW * T + LW, T, (long long)T * T, nullptr, LE, LE, DH);

            softmax_rows<<<dim3(T, NH), 256, 0, stream>>>(
                scoresf, probs_h, word_mask + b * LW, entity_mask + b * LE, scale);

            // ctx = probs @ V : A=[T][T] f16, Bt = Vt[head] = [DH][T]
            gemm_wmma<false, false, half_t><<<dim3(DH / BN, T / BM, NH), 256, 0, stream>>>(
                probs_h, T, (long long)T * T,
                vt_h + (long long)b * H * T, T, (long long)DH * T,
                ctx_h + (long long)b * T * H, H, DH, nullptr, T, DH, T);
        }

        // ---- Self-output + residual LN ----
        gemm_wmma<false, false, float><<<dim3(H / BN, (B * T) / BM, 1), 256, 0, stream>>>(
            ctx_h, H, 0, soT_l, H, 0, sobuf, H, 0, so_b + l * H, B * T, H, H);
        add_res_ln<<<B * T, 256, 0, stream>>>(sobuf, hidden, so_ln_g + l * H,
                                              so_ln_b + l * H, sobuf, so_h);

        // ---- FFN ----
        gemm_wmma<true, false, half_t><<<dim3(FF / BN, (B * T) / BM, 1), 256, 0, stream>>>(
            so_h, H, 0, intT_l, H, 0, ff_h, FF, 0, int_b + l * FF, B * T, FF, H);
        gemm_wmma<false, false, float><<<dim3(H / BN, (B * T) / BM, 1), 256, 0, stream>>>(
            ff_h, FF, 0, outT_l, FF, 0, ff2, H, 0, out_b + l * H, B * T, H, FF);
        add_res_ln<<<B * T, 256, 0, stream>>>(ff2, sobuf, out_ln_g + l * H,
                                              out_ln_b + l * H, hidden, hidden_h);
    }

    copy_out_kernel<<<(B * T * H) / 256, 256, 0, stream>>>(hidden, (float*)d_out);
}
